// NoiseAdaptiveMDTA_25460566130791
// MI455X (gfx1250) — matrix operand order repaired and checked by reading, not compiled, not run on hardware
//
#include <hip/hip_runtime.h>
#include <hip/hip_bf16.h>

typedef __attribute__((ext_vector_type(2)))  float  v2f;
typedef __attribute__((ext_vector_type(8)))  float  v8f;
typedef __attribute__((ext_vector_type(16))) __bf16 v16bf;
typedef __attribute__((ext_vector_type(8)))  __bf16 v8bf;

#define Bb     2
#define Cc     192
#define Hh     256
#define Ww     256
#define NHEADS 4
#define CH     48
#define NPIX   65536
#define C3     576

// ---------------------------------------------------------------------------
// Kernel 0: zero-init atomic accumulators (sumsq + S + temp)
// ---------------------------------------------------------------------------
__global__ void zero_init_kernel(float* __restrict__ p, int n) {
    int i = blockIdx.x * blockDim.x + threadIdx.x;
    if (i < n) p[i] = 0.0f;
}

// ---------------------------------------------------------------------------
// Kernel 1/6: bf16 WMMA GEMM  Y[b][M][NPIX] = W[M][192] @ X[b][192][NPIX]
// fp32 -> bf16 conversion fused into fully static, vectorized LDS staging:
//   A: float4-pair loads along K -> ds_store_b128
//   B: 16 coalesced b32 loads per thread (one column) -> 2x ds_store_b128
//      into a K-transposed tile so fragments are contiguous-K b128 reads.
// K = 192 in chunks of 64 -> 4 v_wmma_f32_16x16x32_bf16 per wave per chunk.
// Row stride 72 bf16 = 144 B = 36 banks: 36*r mod 64 distinct for r=0..15 ->
// conflict-free 16-row fragment reads; rows stay 16B-aligned for b128.
// ---------------------------------------------------------------------------
__global__ __launch_bounds__(256) void gemm192_bf16_wmma(const float* __restrict__ Wm,
                                                         const float* __restrict__ X,
                                                         float* __restrict__ Y, int M) {
    const int K = 192;
    __shared__ alignas(16) __bf16 As[64][72];   // [M-tile][K-chunk]
    __shared__ alignas(16) __bf16 BsT[64][72];  // [N-tile][K-chunk] (transposed)
    int n0 = blockIdx.x * 64;
    int m0 = blockIdx.y * 64;
    int b  = blockIdx.z;
    const float* Xb = X + (size_t)b * K * NPIX;
    float*       Yb = Y + (size_t)b * M * NPIX;
    int tid  = threadIdx.x;
    int wave = tid >> 5;
    int lane = tid & 31;
    int msub  = wave & 3;          // 0..3  (M subtile)
    int nbase = (wave >> 2) * 2;   // 0 or 2 (first of two N subtiles)
    int half = lane >> 4, l16 = lane & 15;
    v8f acc0 = {}, acc1 = {};

    // static staging coordinates
    int am  = tid >> 2;            // 0..63  (A row)
    int aks = (tid & 3) * 16;      // 0,16,32,48 (A k-slot)
    int bn  = tid & 63;            // 0..63  (B column)
    int bkg = (tid >> 6) * 16;     // 0,16,32,48 (B k-group)

    for (int k0 = 0; k0 < K; k0 += 64) {
        __syncthreads();
        // ---- stage A: 64(M) x 64(K), coalesced along K, b128 LDS stores ----
        {
            const float* src = &Wm[(size_t)(m0 + am) * K + k0 + aks];
            #pragma unroll
            for (int g = 0; g < 2; g++) {
                float4 f0 = *(const float4*)(src + g * 8);
                float4 f1 = *(const float4*)(src + g * 8 + 4);
                v8bf v;
                v[0] = (__bf16)f0.x; v[1] = (__bf16)f0.y;
                v[2] = (__bf16)f0.z; v[3] = (__bf16)f0.w;
                v[4] = (__bf16)f1.x; v[5] = (__bf16)f1.y;
                v[6] = (__bf16)f1.z; v[7] = (__bf16)f1.w;
                *(v8bf*)&As[am][aks + g * 8] = v;
            }
        }
        // ---- stage B: 64(K) x 64(N) -> BsT[n][k]; 16 batched b32 loads ----
        {
            float f[16];
            #pragma unroll
            for (int j = 0; j < 16; j++)
                f[j] = Xb[(size_t)(k0 + bkg + j) * NPIX + n0 + bn];
            #pragma unroll
            for (int g = 0; g < 2; g++) {
                v8bf v;
                #pragma unroll
                for (int j = 0; j < 8; j++) v[j] = (__bf16)f[g * 8 + j];
                *(v8bf*)&BsT[bn][bkg + g * 8] = v;
            }
        }
        __syncthreads();
        // ---- two K=32 WMMA steps per chunk ----
        #pragma unroll
        for (int ks = 0; ks < 64; ks += 32) {
            // A frag: lanes 0-15 hold K=ks+0-15, lanes 16-31 K=ks+16-31
            v8bf alo = *(const v8bf*)&As[msub * 16 + l16][ks + half * 16];
            v8bf ahi = *(const v8bf*)&As[msub * 16 + l16][ks + half * 16 + 8];
            v16bf a;
            #pragma unroll
            for (int j = 0; j < 8; j++) { a[j] = alo[j]; a[j + 8] = ahi[j]; }
            // B frag: lanes 0-15 hold K=ks+0-15, lanes 16-31 K=ks+16-31
            v8bf b0lo = *(const v8bf*)&BsT[nbase * 16 + l16][ks + half * 16];
            v8bf b0hi = *(const v8bf*)&BsT[nbase * 16 + l16][ks + half * 16 + 8];
            v8bf b1lo = *(const v8bf*)&BsT[(nbase + 1) * 16 + l16][ks + half * 16];
            v8bf b1hi = *(const v8bf*)&BsT[(nbase + 1) * 16 + l16][ks + half * 16 + 8];
            v16bf bb0, bb1;
            #pragma unroll
            for (int j = 0; j < 8; j++) {
                bb0[j] = b0lo[j]; bb0[j + 8] = b0hi[j];
                bb1[j] = b1lo[j]; bb1[j + 8] = b1hi[j];
            }
            acc0 = __builtin_amdgcn_wmma_f32_16x16x32_bf16(false, a, false, bb0,
                                                           (short)0, acc0, false, false);
            acc1 = __builtin_amdgcn_wmma_f32_16x16x32_bf16(false, a, false, bb1,
                                                           (short)0, acc1, false, false);
        }
    }
    // C/D layout: VGPR i, lanes 0-15 -> M=i ; lanes 16-31 -> M=i+8
    #pragma unroll
    for (int i = 0; i < 8; i++) {
        int row = m0 + msub * 16 + i + half * 8;
        Yb[(size_t)row * NPIX + n0 + nbase * 16 + l16]       = acc0[i];
        Yb[(size_t)row * NPIX + n0 + (nbase + 1) * 16 + l16] = acc1[i];
    }
}

// ---------------------------------------------------------------------------
// Kernel 2: depthwise 3x3 SAME conv + fused sum-of-squares for q,k channels
// grid (64 pixel-chunks, 576 channels, 2 batches), 256 thr, 4 px/thread
// ---------------------------------------------------------------------------
__global__ __launch_bounds__(256) void dwconv3x3_kernel(const float* __restrict__ in,
                                                        const float* __restrict__ wdw,
                                                        float* __restrict__ out,
                                                        float* __restrict__ sumsq) {
    int chunk = blockIdx.x;
    int c     = blockIdx.y;
    int b     = blockIdx.z;
    const float* src = in  + ((size_t)b * C3 + c) * NPIX;
    float*       dst = out + ((size_t)b * C3 + c) * NPIX;
    float w[9];
    #pragma unroll
    for (int i = 0; i < 9; i++) w[i] = wdw[c * 9 + i];

    int base = chunk * 1024 + threadIdx.x * 4;
    float local = 0.0f;
    #pragma unroll
    for (int j = 0; j < 4; j++) {
        int p = base + j;
        int y = p >> 8;
        int x = p & 255;
        float acc = 0.0f;
        #pragma unroll
        for (int dy = -1; dy <= 1; dy++) {
            int yy = y + dy;
            if (yy < 0 || yy >= Hh) continue;
            #pragma unroll
            for (int dx = -1; dx <= 1; dx++) {
                int xx = x + dx;
                if (xx < 0 || xx >= Ww) continue;
                acc += src[yy * Ww + xx] * w[(dy + 1) * 3 + (dx + 1)];
            }
        }
        dst[p] = acc;
        local += acc * acc;
    }
    if (c < 2 * Cc) {  // q and k channels feed the L2 norms
        __shared__ float red[256];
        red[threadIdx.x] = local;
        __syncthreads();
        for (int s = 128; s > 0; s >>= 1) {
            if (threadIdx.x < s) red[threadIdx.x] += red[threadIdx.x + s];
            __syncthreads();
        }
        if (threadIdx.x == 0) atomicAdd(&sumsq[(size_t)b * (2 * Cc) + c], red[0]);
    }
}

// ---------------------------------------------------------------------------
// Kernel 3: noise MLP -> temp[b][head] = base_temp[h] * (2 - sigmoid(mlp))
// ---------------------------------------------------------------------------
__global__ void noise_mlp_kernel(const float* __restrict__ noise,
                                 const float* __restrict__ base_temp,
                                 const float* __restrict__ w1,
                                 const float* __restrict__ w2,
                                 float* __restrict__ temp) {
    __shared__ float nm1[Bb][CH];
    int b = threadIdx.x >> 6;  // 128 threads: 64 per batch
    int j = threadIdx.x & 63;
    if (j < CH) {
        float v = w1[j] * noise[b];
        nm1[b][j] = (v >= 0.0f) ? v : 0.2f * v;  // LeakyReLU(0.2)
    }
    __syncthreads();
    if (j < NHEADS) {
        float acc = 0.0f;
        for (int i = 0; i < CH; i++) acc += w2[j * CH + i] * nm1[b][i];
        float sig = 1.0f / (1.0f + __expf(-acc));
        temp[b * NHEADS + j] = base_temp[j] * (2.0f - sig);
    }
}

// ---------------------------------------------------------------------------
// Kernel 4: S[b,h][48][48] += q_raw @ k_raw^T  (split-K fp32 WMMA, atomics)
// 9 waves = the 3x3 grid of 16x16 tiles; grid (64 k-splits, 8 b*h)
// ---------------------------------------------------------------------------
__global__ __launch_bounds__(288) void qk_gemm_wmma(const float* __restrict__ qkv,
                                                    float* __restrict__ S) {
    __shared__ float qs[CH][65];
    __shared__ float ks[CH][65];
    int ksplit = blockIdx.x;        // 0..63
    int bh     = blockIdx.y;        // 0..7
    int b = bh >> 2, h = bh & 3;
    const float* qbase = qkv + ((size_t)b * C3 + h * CH) * NPIX;
    const float* kbase = qkv + ((size_t)b * C3 + Cc + h * CH) * NPIX;
    int tid  = threadIdx.x;
    int wave = tid >> 5;            // 0..8
    int lane = tid & 31;
    int tm = wave / 3, tn = wave % 3;
    int half = lane >> 4, l16 = lane & 15;
    v8f acc = {};
    int kstart = ksplit * (NPIX / 64);  // 1024 K-values per block

    for (int kc = 0; kc < 1024; kc += 64) {
        __syncthreads();
        for (int i = tid; i < CH * 64; i += 288) {
            int r = i >> 6, c = i & 63;
            qs[r][c] = qbase[(size_t)r * NPIX + kstart + kc + c];
            ks[r][c] = kbase[(size_t)r * NPIX + kstart + kc + c];
        }
        __syncthreads();
        #pragma unroll
        for (int kk = 0; kk < 64; kk += 4) {
            v2f a, bb;
            a.x  = qs[tm * 16 + l16][kk + half * 2 + 0];
            a.y  = qs[tm * 16 + l16][kk + half * 2 + 1];
            bb.x = ks[tn * 16 + l16][kk + half * 2 + 0];  // B = k^T
            bb.y = ks[tn * 16 + l16][kk + half * 2 + 1];
            acc = __builtin_amdgcn_wmma_f32_16x16x4_f32(false, a, false, bb,
                                                        (short)0, acc, false, false);
        }
    }
    float* Sb = S + (size_t)bh * CH * CH;
    #pragma unroll
    for (int i = 0; i < 8; i++) {
        int row = tm * 16 + i + half * 8;
        int col = tn * 16 + l16;
        atomicAdd(&Sb[row * CH + col], acc[i]);
    }
}

// ---------------------------------------------------------------------------
// Kernel 5: scale by 1/(|q||k|) * temp, softmax over last dim (48), in place
// ---------------------------------------------------------------------------
__global__ void softmax48_kernel(float* __restrict__ S,
                                 const float* __restrict__ sumsq,
                                 const float* __restrict__ temp) {
    int bh = blockIdx.x;
    int b = bh >> 2, h = bh & 3;
    __shared__ float nk[CH];
    int t = threadIdx.x;  // 64 threads
    if (t < CH) nk[t] = fmaxf(sqrtf(sumsq[b * (2 * Cc) + Cc + h * CH + t]), 1e-12f);
    __syncthreads();
    if (t < CH) {
        float nq  = fmaxf(sqrtf(sumsq[b * (2 * Cc) + h * CH + t]), 1e-12f);
        float tmp = temp[bh];
        float* row = S + ((size_t)bh * CH + t) * CH;
        float vals[CH];
        float mx = -1e30f;
        for (int d = 0; d < CH; d++) {
            float s = row[d] / (nq * nk[d]) * tmp;
            vals[d] = s;
            mx = fmaxf(mx, s);
        }
        float sum = 0.0f;
        for (int d = 0; d < CH; d++) { vals[d] = __expf(vals[d] - mx); sum += vals[d]; }
        float inv = 1.0f / sum;
        for (int d = 0; d < CH; d++) row[d] = vals[d] * inv;
    }
}

// ---------------------------------------------------------------------------
// Kernel 6: out_pre[b, h*48+m, n] = attn[b,h] @ v[b,h]   (fp32 WMMA)
// 8 waves: wave w owns N-subtile w (16 wide), 3 M-subtiles; N block tile 128
// ---------------------------------------------------------------------------
__global__ __launch_bounds__(256) void av_gemm_wmma(const float* __restrict__ attn,
                                                    const float* __restrict__ qkv,
                                                    float* __restrict__ outp) {
    __shared__ float as[CH][49];
    __shared__ float vs[CH][129];
    int nb = blockIdx.x;   // 0..511
    int bh = blockIdx.y;   // 0..7
    int b = bh >> 2, h = bh & 3;
    const float* A = attn + (size_t)bh * CH * CH;
    const float* V = qkv + ((size_t)b * C3 + 2 * Cc + h * CH) * NPIX;
    float*       O = outp + ((size_t)b * Cc + h * CH) * NPIX;
    int tid = threadIdx.x;
    int n0  = nb * 128;
    for (int i = tid; i < CH * CH; i += 256) as[i / CH][i % CH] = A[i];
    for (int i = tid; i < CH * 128; i += 256) {
        int r = i >> 7, c = i & 127;
        vs[r][c] = V[(size_t)r * NPIX + n0 + c];
    }
    __syncthreads();
    int wave = tid >> 5, lane = tid & 31;
    int half = lane >> 4, l16 = lane & 15;
    v8f acc[3] = {};
    #pragma unroll
    for (int kk = 0; kk < CH; kk += 4) {
        v2f bb;
        bb.x = vs[kk + half * 2 + 0][wave * 16 + l16];
        bb.y = vs[kk + half * 2 + 1][wave * 16 + l16];
        #pragma unroll
        for (int m = 0; m < 3; m++) {
            v2f a;
            a.x = as[m * 16 + l16][kk + half * 2 + 0];
            a.y = as[m * 16 + l16][kk + half * 2 + 1];
            acc[m] = __builtin_amdgcn_wmma_f32_16x16x4_f32(false, a, false, bb,
                                                           (short)0, acc[m], false, false);
        }
    }
    #pragma unroll
    for (int m = 0; m < 3; m++)
        #pragma unroll
        for (int i = 0; i < 8; i++) {
            int row = m * 16 + i + half * 8;
            O[(size_t)row * NPIX + n0 + wave * 16 + l16] = acc[m][i];
        }
}

// ---------------------------------------------------------------------------
extern "C" void kernel_launch(void* const* d_in, const int* in_sizes, int n_in,
                              void* d_out, int out_size, void* d_ws, size_t ws_size,
                              hipStream_t stream) {
    const float* x      = (const float*)d_in[0];
    const float* noise  = (const float*)d_in[1];
    const float* base_t = (const float*)d_in[2];
    const float* w_nm1  = (const float*)d_in[3];
    const float* w_nm2  = (const float*)d_in[4];
    const float* w_qkv  = (const float*)d_in[5];
    const float* w_dw   = (const float*)d_in[6];
    const float* w_out  = (const float*)d_in[7];
    float* out = (float*)d_out;
    float* ws  = (float*)d_ws;

    const size_t QKV_ELEMS = (size_t)Bb * C3 * NPIX;       // 75,497,472
    float* qkv_pre = ws;                                   // [2][576][65536]
    float* qkv_dw  = ws + QKV_ELEMS;                       // [2][576][65536]
    float* sumsq   = ws + 2 * QKV_ELEMS;                   // [2][384]
    float* S       = sumsq + Bb * 2 * Cc;                  // [2][4][48][48]
    float* temp    = S + (size_t)Bb * NHEADS * CH * CH;    // [2][4]
    float* out_pre = qkv_pre;                              // reuse (qkv_pre dead)

    const int nzero = Bb * 2 * Cc + Bb * NHEADS * CH * CH + Bb * NHEADS;
    zero_init_kernel<<<(nzero + 255) / 256, 256, 0, stream>>>(sumsq, nzero);

    // qkv = W_qkv @ x  (bf16 WMMA GEMM, M=576)
    gemm192_bf16_wmma<<<dim3(NPIX / 64, C3 / 64, Bb), 256, 0, stream>>>(w_qkv, x, qkv_pre, C3);

    // depthwise 3x3 + sumsq for q,k
    dwconv3x3_kernel<<<dim3(64, C3, Bb), 256, 0, stream>>>(qkv_pre, w_dw, qkv_dw, sumsq);

    // noise MLP -> temperatures
    noise_mlp_kernel<<<1, 128, 0, stream>>>(noise, base_t, w_nm1, w_nm2, temp);

    // S = q @ k^T (split-K fp32 WMMA + atomics)
    qk_gemm_wmma<<<dim3(64, Bb * NHEADS), 288, 0, stream>>>(qkv_dw, S);

    // softmax with norm/temp scaling (in place)
    softmax48_kernel<<<Bb * NHEADS, 64, 0, stream>>>(S, sumsq, temp);

    // out_pre = attn @ v (fp32 WMMA)
    av_gemm_wmma<<<dim3(NPIX / 128, Bb * NHEADS), 256, 0, stream>>>(S, qkv_dw, out_pre);

    // final 1x1 conv: out = w_out @ out_pre (bf16 WMMA GEMM, M=192)
    gemm192_bf16_wmma<<<dim3(NPIX / 64, Cc / 64, Bb), 256, 0, stream>>>(w_out, out_pre, out, Cc);
}
